// DualGPDClassifier_23175643529716
// MI455X (gfx1250) — matrix operand order repaired and checked by reading, not compiled, not run on hardware
//
#include <hip/hip_runtime.h>
#include <hip/hip_bf16.h>

// ---------------------------------------------------------------------------
// PointNet++-style DualGPDClassifier for MI455X (gfx1250, wave32, WMMA).
//
// Pipeline (all on `stream`, graph-capture safe, no mallocs):
//   build xyz -> FPS -> ball query -> group -> [WMMA GEMM -> colstats -> BN+act]x3
//   -> maxpool  (x3 SA stages) -> head via the same WMMA GEMM + BN machinery.
//
// Precision: activations f16 (halves HBM traffic; BN renormalizes each layer),
// GEMM accumulate + BN stats in f32 via v_wmma_f32_16x16x32_f16.
//
// Input flattening assumption (setup_inputs insertion order):
//   [0] x1, [1] x2,
//   sa1: [2..5]=W1,b1,g1,bt1 [6..9]=L2 [10..13]=L3
//   sa2: [14..17] [18..21] [22..25]
//   sa3: [26..29] [30..33] [34..37]
//   head: [38]fc1_w [39]fc1_b [40]bn1_g [41]bn1_b [42]fc2_w [43]fc2_b
//         [44]bn2_g [45]bn2_b [46]fc3a_w [47]bn3a_g [48]bn3a_b
//         [49]fc3b_w [50]bn3b_g [51]bn3b_b [52]fc3c_w [53]bn3c_g [54]bn3c_b
//         [55]fc3d_w [56]fc3d_b
// ---------------------------------------------------------------------------

typedef __attribute__((ext_vector_type(16))) _Float16 v16h;
typedef __attribute__((ext_vector_type(8)))  _Float16 v8h;
typedef __attribute__((ext_vector_type(8)))  float    v8f;

#define NB   16      // batch
#define NPTS 4096    // points per cloud
#define S1   512
#define NS1  32
#define S2   128
#define NS2  64
#define EPSF 1e-5f

#define CDIV(a, b) (((a) + (b) - 1) / (b))

// ---------------------------------------------------------------- utilities

__global__ void zero_kernel(float* p, int n) {
    int g = blockIdx.x * blockDim.x + threadIdx.x;
    if (g < n) p[g] = 0.0f;
}

// W [K,N] f32 -> Wt [N,Kpad] f16, zero-padded K..Kpad-1
__global__ void prep_weight_kernel(const float* __restrict__ W, _Float16* __restrict__ Wt,
                                   int K, int N, int Kpad) {
    int g = blockIdx.x * blockDim.x + threadIdx.x;
    int total = N * Kpad;
    if (g >= total) return;
    int n = g / Kpad, k = g % Kpad;
    Wt[g] = (k < K) ? (_Float16)W[(size_t)k * N + n] : (_Float16)0.0f;
}

// x1,x2 [B,3,2048] -> xyz [B,4096,3]
__global__ void build_xyz_kernel(const float* __restrict__ x1, const float* __restrict__ x2,
                                 float* __restrict__ xyz) {
    int g = blockIdx.x * blockDim.x + threadIdx.x;
    if (g >= NB * NPTS) return;
    int b = g / NPTS, n = g % NPTS;
    const float* src = (n < 2048) ? x1 : x2;
    int nn = (n < 2048) ? n : n - 2048;
    xyz[(size_t)g * 3 + 0] = src[((size_t)b * 3 + 0) * 2048 + nn];
    xyz[(size_t)g * 3 + 1] = src[((size_t)b * 3 + 1) * 2048 + nn];
    xyz[(size_t)g * 3 + 2] = src[((size_t)b * 3 + 2) * 2048 + nn];
}

// ---------------------------------------------------------------- FPS

// One block per batch, 1024 threads, dist[] in LDS. Sequential over npoint,
// block-wide argmax each step (tie -> lowest index, mimicking jnp.argmax).
__global__ void fps_kernel(const float* __restrict__ xyz, int N, int npoint,
                           int* __restrict__ fidx) {
    __shared__ float dist[NPTS];
    __shared__ float rmax[1024];
    __shared__ int   ridx[1024];
    __shared__ float cpt[3];
    __shared__ int   farS;
    int b = blockIdx.x, t = threadIdx.x, bt = blockDim.x;
    const float* base = xyz + (size_t)b * N * 3;
    for (int i = t; i < N; i += bt) dist[i] = 1e10f;
    if (t == 0) farS = 0;
    __syncthreads();
    for (int it = 0; it < npoint; ++it) {
        if (t == 0) {
            int f = farS;
            fidx[b * npoint + it] = f;
            cpt[0] = base[f * 3 + 0]; cpt[1] = base[f * 3 + 1]; cpt[2] = base[f * 3 + 2];
        }
        __syncthreads();
        float cx = cpt[0], cy = cpt[1], cz = cpt[2];
        float bm = -1.0f; int bi = 0;
        for (int i = t; i < N; i += bt) {
            float dx = base[i * 3 + 0] - cx;
            float dy = base[i * 3 + 1] - cy;
            float dz = base[i * 3 + 2] - cz;
            float d = dx * dx + dy * dy + dz * dz;
            float o = dist[i];
            d = fminf(o, d);
            dist[i] = d;
            if (d > bm) { bm = d; bi = i; }
        }
        rmax[t] = bm; ridx[t] = bi;
        __syncthreads();
        for (int s = bt >> 1; s > 0; s >>= 1) {
            if (t < s) {
                if (rmax[t + s] > rmax[t] ||
                    (rmax[t + s] == rmax[t] && ridx[t + s] < ridx[t])) {
                    rmax[t] = rmax[t + s]; ridx[t] = ridx[t + s];
                }
            }
            __syncthreads();
        }
        if (t == 0) farS = ridx[0];
        __syncthreads();
    }
}

// ctr[q] = xyz[b, fidx[q]]
__global__ void gather_centroids_kernel(const float* __restrict__ xyz, const int* __restrict__ fidx,
                                        float* __restrict__ ctr, int N, int S, int total) {
    int q = blockIdx.x * blockDim.x + threadIdx.x;
    if (q >= total) return;
    int b = q / S;
    int f = fidx[q];
    const float* p = xyz + ((size_t)b * N + f) * 3;
    ctr[(size_t)q * 3 + 0] = p[0];
    ctr[(size_t)q * 3 + 1] = p[1];
    ctr[(size_t)q * 3 + 2] = p[2];
}

// ---------------------------------------------------------------- ball query

// First `ns` indices (ascending) with d2 <= r2; pad with first hit.
__global__ void ball_query_kernel(const float* __restrict__ xyz, const float* __restrict__ ctr,
                                  int* __restrict__ idx, int N, int S, int ns, float r2, int total) {
    int q = blockIdx.x * blockDim.x + threadIdx.x;
    if (q >= total) return;
    int b = q / S;
    const float* base = xyz + (size_t)b * N * 3;
    float cx = ctr[(size_t)q * 3 + 0];
    float cy = ctr[(size_t)q * 3 + 1];
    float cz = ctr[(size_t)q * 3 + 2];
    int* o = idx + (size_t)q * ns;
    int cnt = 0;
    for (int i = 0; i < N && cnt < ns; ++i) {
        float dx = base[i * 3 + 0] - cx;
        float dy = base[i * 3 + 1] - cy;
        float dz = base[i * 3 + 2] - cz;
        if (dx * dx + dy * dy + dz * dz <= r2) o[cnt++] = i;
    }
    int f = (cnt > 0) ? o[0] : 0;
    for (; cnt < ns; ++cnt) o[cnt] = f;
}

// ---------------------------------------------------------------- grouping

// SA1: A[g, 0..2] = xyz[idx]-ctr, pad to Kpad with zeros
__global__ void group_xyz_kernel(const float* __restrict__ xyz, const float* __restrict__ ctr,
                                 const int* __restrict__ idx, _Float16* __restrict__ A,
                                 int N, int S, int ns, int Kpad, int total) {
    int g = blockIdx.x * blockDim.x + threadIdx.x;
    if (g >= total) return;
    int k = g % ns, q = g / ns, b = q / S;
    int pi = idx[(size_t)q * ns + k];
    const float* p = xyz + ((size_t)b * N + pi) * 3;
    _Float16* row = A + (size_t)g * Kpad;
    row[0] = (_Float16)(p[0] - ctr[(size_t)q * 3 + 0]);
    row[1] = (_Float16)(p[1] - ctr[(size_t)q * 3 + 1]);
    row[2] = (_Float16)(p[2] - ctr[(size_t)q * 3 + 2]);
    for (int c = 3; c < Kpad; ++c) row[c] = (_Float16)0.0f;
}

// SA2: A[g] = concat(xyz[idx]-ctr (3), pts[idx] (C)), pad to Kpad
__global__ void group_xyz_pts_kernel(const float* __restrict__ xyz, const float* __restrict__ ctr,
                                     const int* __restrict__ idx, const _Float16* __restrict__ pts,
                                     _Float16* __restrict__ A, int Nsrc, int S, int ns, int C,
                                     int Kpad, int total) {
    int g = blockIdx.x * blockDim.x + threadIdx.x;
    if (g >= total) return;
    int k = g % ns, q = g / ns, b = q / S;
    int pi = idx[(size_t)q * ns + k];
    const float* p = xyz + ((size_t)b * Nsrc + pi) * 3;
    _Float16* row = A + (size_t)g * Kpad;
    row[0] = (_Float16)(p[0] - ctr[(size_t)q * 3 + 0]);
    row[1] = (_Float16)(p[1] - ctr[(size_t)q * 3 + 1]);
    row[2] = (_Float16)(p[2] - ctr[(size_t)q * 3 + 2]);
    const _Float16* pp = pts + ((size_t)b * Nsrc + pi) * C;
    for (int c = 0; c < C; ++c) row[3 + c] = pp[c];
    for (int c = 3 + C; c < Kpad; ++c) row[c] = (_Float16)0.0f;
}

// SA3 input rows: cols 0..2 = l2_xyz, cols 259..Kpad-1 = 0 (cols 3..258 by maxpool)
__global__ void sa3_fill_kernel(const float* __restrict__ ctr, _Float16* __restrict__ A,
                                int rows, int C, int Kpad) {
    int g = blockIdx.x * blockDim.x + threadIdx.x;
    if (g >= rows) return;
    _Float16* row = A + (size_t)g * Kpad;
    row[0] = (_Float16)ctr[(size_t)g * 3 + 0];
    row[1] = (_Float16)ctr[(size_t)g * 3 + 1];
    row[2] = (_Float16)ctr[(size_t)g * 3 + 2];
    for (int c = C; c < Kpad; ++c) row[c] = (_Float16)0.0f;
}

// ---------------------------------------------------------------- WMMA GEMM

// Y[M,N] = A[M,K] @ Wt^T + bias.  A: [M,K] f16 row-major (K % 32 == 0).
// Wt: [N,K] f16 row-major (pre-transposed weights). One wave per 16x16 tile.
// Fragment layouts per CDNA5 ISA 7.12.2:
//   A (16x32 f16): lanes 0-15 row=lane, elems {K0..7, K16..23};
//                  lanes 16-31 same rows, elems {K8..15, K24..31}.
//   B (32x16 f16): lanes 0-15 col=lane, elems {K0..7, K16..23}; lanes 16-31 {K8..15, K24..31}.
//   C/D (16x16 f32): lanes 0-15 rows 0..7 (vgpr r), lanes 16-31 rows 8..15.
__global__ void wmma_gemm_kernel(const _Float16* __restrict__ A, const _Float16* __restrict__ Wt,
                                 const float* __restrict__ bias, float* __restrict__ Y,
                                 int M, int N, int K) {
    int wave = blockIdx.x * (blockDim.x >> 5) + (threadIdx.x >> 5);
    int tilesN = N >> 4;
    int totalTiles = (M >> 4) * tilesN;
    if (wave >= totalTiles) return;           // wave-uniform: EXEC stays all-ones
    int tm = wave / tilesN, tn = wave % tilesN;
    int lane = threadIdx.x & 31;
    int l = lane & 15;
    int half = lane >> 4;
    const _Float16* aRow = A + (size_t)(tm * 16 + l) * K;
    const _Float16* bRow = Wt + (size_t)(tn * 16 + l) * K;
    v8f acc = {};
    for (int k = 0; k < K; k += 32) {
        if (k + 64 < K) {                     // uniform branch: prefetch 2 steps ahead
            __builtin_prefetch(aRow + k + 64, 0, 1);
            __builtin_prefetch(bRow + k + 64, 0, 1);
        }
        v8h a0 = *(const v8h*)(aRow + k + half * 8);
        v8h a1 = *(const v8h*)(aRow + k + half * 8 + 16);
        v16h bv = __builtin_shufflevector(*(const v8h*)(bRow + k + half * 8), *(const v8h*)(bRow + k + half * 8 + 16), 0, 1, 2, 3, 4, 5, 6, 7, 8, 9, 10, 11, 12, 13, 14, 15);
        v16h a;
#pragma unroll
        for (int e = 0; e < 8; ++e) { a[e] = a0[e]; a[e + 8] = a1[e]; }
        acc = __builtin_amdgcn_wmma_f32_16x16x32_f16(false, a, false, bv,
                                                     (short)0, acc, false, false);
    }
    int col = tn * 16 + l;
    float bb = bias ? bias[col] : 0.0f;
    int rbase = tm * 16 + half * 8;
#pragma unroll
    for (int r = 0; r < 8; ++r)
        Y[(size_t)(rbase + r) * N + col] = acc[r] + bb;
}

// ---------------------------------------------------------------- BN machinery

#define MAXC 1024

// stats[0..N) += col sums, stats[N..2N) += col sums of squares
__global__ void colstats_kernel(const float* __restrict__ Y, float* __restrict__ stats,
                                int M, int N, int rowsPerBlock) {
    __shared__ float s[2 * MAXC];
    int t = threadIdx.x;
    for (int i = t; i < 2 * N; i += blockDim.x) s[i] = 0.0f;
    __syncthreads();
    int r0 = blockIdx.x * rowsPerBlock;
    int r1 = min(M, r0 + rowsPerBlock);
    int cnt = (r1 - r0) * N;
    for (int i = t; i < cnt; i += blockDim.x) {
        int r = r0 + i / N;
        int c = i - (i / N) * N;
        float v = Y[(size_t)r * N + c];
        atomicAdd(&s[c], v);
        atomicAdd(&s[N + c], v * v);
    }
    __syncthreads();
    for (int i = t; i < 2 * N; i += blockDim.x) atomicAdd(&stats[i], s[i]);
}

// act: 0 = relu, 1 = elu(alpha=1). Writes f16 into [M,Kpad], zero-padding cols N..Kpad-1.
__global__ void bn_act_f16_kernel(const float* __restrict__ Y, const float* __restrict__ stats,
                                  const float* __restrict__ gam, const float* __restrict__ bet,
                                  _Float16* __restrict__ X, int M, int N, int Kpad,
                                  float invCnt, int act) {
    size_t idx = (size_t)blockIdx.x * blockDim.x + threadIdx.x;
    size_t total = (size_t)M * Kpad;
    if (idx >= total) return;
    int c = (int)(idx % Kpad);
    size_t r = idx / Kpad;
    if (c >= N) { X[idx] = (_Float16)0.0f; return; }
    float v = Y[r * N + c];
    float mu = stats[c] * invCnt;
    float var = stats[N + c] * invCnt - mu * mu;
    v = (v - mu) * rsqrtf(var + EPSF) * gam[c] + bet[c];
    if (act == 0)      v = fmaxf(v, 0.0f);
    else if (act == 1) v = (v > 0.0f) ? v : (expf(v) - 1.0f);
    X[idx] = (_Float16)v;
}

// max over group dim: X [BS*Kg, C] -> out[bs*ostride + ooff + c]
__global__ void maxpool_kernel(const _Float16* __restrict__ X, _Float16* __restrict__ out,
                               int BS, int Kg, int C, int ostride, int ooff) {
    int g = blockIdx.x * blockDim.x + threadIdx.x;
    if (g >= BS * C) return;
    int bs = g / C, c = g % C;
    const _Float16* p = X + (size_t)bs * Kg * C + c;
    float m = -3.4e38f;
    for (int k = 0; k < Kg; ++k) m = fmaxf(m, (float)p[(size_t)k * C]);
    out[(size_t)bs * ostride + ooff + c] = (_Float16)m;
}

// ---------------------------------------------------------------- head tail

__global__ void head_final_kernel(const _Float16* __restrict__ X, const float* __restrict__ Wf,
                                  const float* __restrict__ bf, float* __restrict__ out) {
    int r = threadIdx.x;
    if (r >= 16) return;
    float acc = bf[0];
    for (int k = 0; k < 128; ++k) acc += (float)X[r * 128 + k] * Wf[k];
    out[r] = 1.0f / (1.0f + expf(-acc));
}

// ---------------------------------------------------------------- launch

extern "C" void kernel_launch(void* const* d_in, const int* in_sizes, int n_in,
                              void* d_out, int out_size, void* d_ws, size_t ws_size,
                              hipStream_t stream) {
    if (n_in < 57) return;

    // ---- workspace bump allocator (~250 MB total) ----
    char* wsb = (char*)d_ws;
    size_t off = 0;
    auto alloc = [&](size_t bytes) -> void* {
        void* p = wsb + off;
        off = (off + bytes + 255) & ~(size_t)255;
        return p;
    };

    const int M1 = NB * S1 * NS1;   // 262144
    const int M2 = NB * S2 * NS2;   // 131072
    const int M3 = NB * S2;         // 2048

    float*     xyz   = (float*)alloc((size_t)NB * NPTS * 3 * 4);
    int*       fidx1 = (int*)alloc((size_t)NB * S1 * 4);
    float*     ctr1  = (float*)alloc((size_t)NB * S1 * 3 * 4);
    int*       idx1  = (int*)alloc((size_t)NB * S1 * NS1 * 4);
    _Float16*  l1pts = (_Float16*)alloc((size_t)NB * S1 * 128 * 2);
    int*       fidx2 = (int*)alloc((size_t)NB * S2 * 4);
    float*     ctr2  = (float*)alloc((size_t)NB * S2 * 3 * 4);
    int*       idx2  = (int*)alloc((size_t)NB * S2 * NS2 * 4);
    _Float16*  sa3in = (_Float16*)alloc((size_t)M3 * 288 * 2);
    _Float16*  Xh0   = (_Float16*)alloc((size_t)16 * 1024 * 2);
    _Float16*  Xh1   = (_Float16*)alloc((size_t)16 * 512 * 2);
    _Float16*  Xh2   = (_Float16*)alloc((size_t)16 * 512 * 2);
    _Float16*  Xh3   = (_Float16*)alloc((size_t)16 * 256 * 2);
    _Float16*  Xh4   = (_Float16*)alloc((size_t)16 * 256 * 2);
    _Float16*  Xh5   = (_Float16*)alloc((size_t)16 * 128 * 2);
    float*     stats = (float*)alloc((size_t)2 * MAXC * 4);

    // transposed + K-padded f16 weights
    _Float16* wt_s1l1 = (_Float16*)alloc((size_t)64 * 32 * 2);
    _Float16* wt_s1l2 = (_Float16*)alloc((size_t)64 * 64 * 2);
    _Float16* wt_s1l3 = (_Float16*)alloc((size_t)128 * 64 * 2);
    _Float16* wt_s2l1 = (_Float16*)alloc((size_t)128 * 160 * 2);
    _Float16* wt_s2l2 = (_Float16*)alloc((size_t)128 * 128 * 2);
    _Float16* wt_s2l3 = (_Float16*)alloc((size_t)256 * 128 * 2);
    _Float16* wt_s3l1 = (_Float16*)alloc((size_t)256 * 288 * 2);
    _Float16* wt_s3l2 = (_Float16*)alloc((size_t)512 * 256 * 2);
    _Float16* wt_s3l3 = (_Float16*)alloc((size_t)1024 * 512 * 2);
    _Float16* wt_h1   = (_Float16*)alloc((size_t)512 * 1024 * 2);
    _Float16* wt_h2   = (_Float16*)alloc((size_t)512 * 512 * 2);
    _Float16* wt_h3a  = (_Float16*)alloc((size_t)256 * 512 * 2);
    _Float16* wt_h3b  = (_Float16*)alloc((size_t)256 * 256 * 2);
    _Float16* wt_h3c  = (_Float16*)alloc((size_t)128 * 256 * 2);

    _Float16* XA = (_Float16*)alloc((size_t)M2 * 160 * 2);   // 40 MiB max user
    _Float16* XB = (_Float16*)alloc((size_t)M1 * 128 * 2);   // 64 MiB max user
    float*    Yb = (float*)alloc((size_t)M1 * 128 * 4);      // 128 MiB f32 GEMM out

    const float* fin[64];
    for (int i = 0; i < n_in && i < 64; ++i) fin[i] = (const float*)d_in[i];

    auto prep = [&](int wI, _Float16* dst, int K, int N, int Kpad) {
        int total = N * Kpad;
        prep_weight_kernel<<<CDIV(total, 256), 256, 0, stream>>>(fin[wI], dst, K, N, Kpad);
    };
    auto gemm = [&](const _Float16* A, const _Float16* Wt, const float* bias,
                    float* Y, int M, int N, int K) {
        int tiles = (M / 16) * (N / 16);
        wmma_gemm_kernel<<<CDIV(tiles, 4), 128, 0, stream>>>(A, Wt, bias, Y, M, N, K);
    };
    auto bnblock = [&](float* Y, int M, int N, const float* g, const float* bt,
                       _Float16* Xout, int Kpad, int act) {
        zero_kernel<<<CDIV(2 * N, 256), 256, 0, stream>>>(stats, 2 * N);
        int rpb = CDIV(M, 2048); if (rpb < 1) rpb = 1;
        colstats_kernel<<<CDIV(M, rpb), 256, 0, stream>>>(Y, stats, M, N, rpb);
        size_t tot = (size_t)M * Kpad;
        bn_act_f16_kernel<<<(unsigned)CDIV(tot, 256), 256, 0, stream>>>(
            Y, stats, g, bt, Xout, M, N, Kpad, 1.0f / (float)M, act);
    };

    // ---- weight prep ----
    prep(2,  wt_s1l1,   3,   64,  32);
    prep(6,  wt_s1l2,  64,   64,  64);
    prep(10, wt_s1l3,  64,  128,  64);
    prep(14, wt_s2l1, 131,  128, 160);
    prep(18, wt_s2l2, 128,  128, 128);
    prep(22, wt_s2l3, 128,  256, 128);
    prep(26, wt_s3l1, 259,  256, 288);
    prep(30, wt_s3l2, 256,  512, 256);
    prep(34, wt_s3l3, 512, 1024, 512);
    prep(38, wt_h1,  1024,  512, 1024);
    prep(42, wt_h2,   512,  512, 512);
    prep(46, wt_h3a,  512,  256, 512);
    prep(49, wt_h3b,  256,  256, 256);
    prep(52, wt_h3c,  256,  128, 256);

    // ---- build point cloud ----
    build_xyz_kernel<<<CDIV(NB * NPTS, 256), 256, 0, stream>>>(fin[0], fin[1], xyz);

    // ---- SA1 ----
    fps_kernel<<<NB, 1024, 0, stream>>>(xyz, NPTS, S1, fidx1);
    gather_centroids_kernel<<<CDIV(NB * S1, 256), 256, 0, stream>>>(xyz, fidx1, ctr1, NPTS, S1, NB * S1);
    ball_query_kernel<<<CDIV(NB * S1, 256), 256, 0, stream>>>(xyz, ctr1, idx1, NPTS, S1, NS1, 0.04f, NB * S1);
    group_xyz_kernel<<<CDIV(M1, 256), 256, 0, stream>>>(xyz, ctr1, idx1, XA, NPTS, S1, NS1, 32, M1);

    gemm(XA, wt_s1l1, fin[3], Yb, M1, 64, 32);
    bnblock(Yb, M1, 64, fin[4], fin[5], XB, 64, 0);
    gemm(XB, wt_s1l2, fin[7], Yb, M1, 64, 64);
    bnblock(Yb, M1, 64, fin[8], fin[9], XA, 64, 0);
    gemm(XA, wt_s1l3, fin[11], Yb, M1, 128, 64);
    bnblock(Yb, M1, 128, fin[12], fin[13], XB, 128, 0);
    maxpool_kernel<<<CDIV(NB * S1 * 128, 256), 256, 0, stream>>>(XB, l1pts, NB * S1, NS1, 128, 128, 0);

    // ---- SA2 ----
    fps_kernel<<<NB, 1024, 0, stream>>>(ctr1, S1, S2, fidx2);
    gather_centroids_kernel<<<CDIV(NB * S2, 256), 256, 0, stream>>>(ctr1, fidx2, ctr2, S1, S2, NB * S2);
    ball_query_kernel<<<CDIV(NB * S2, 256), 256, 0, stream>>>(ctr1, ctr2, idx2, S1, S2, NS2, 0.16f, NB * S2);
    group_xyz_pts_kernel<<<CDIV(M2, 256), 256, 0, stream>>>(ctr1, ctr2, idx2, l1pts, XA, S1, S2, NS2, 128, 160, M2);

    gemm(XA, wt_s2l1, fin[15], Yb, M2, 128, 160);
    bnblock(Yb, M2, 128, fin[16], fin[17], XB, 128, 0);
    gemm(XB, wt_s2l2, fin[19], Yb, M2, 128, 128);
    bnblock(Yb, M2, 128, fin[20], fin[21], XA, 128, 0);
    gemm(XA, wt_s2l3, fin[23], Yb, M2, 256, 128);
    bnblock(Yb, M2, 256, fin[24], fin[25], XB, 256, 0);

    sa3_fill_kernel<<<CDIV(M3, 256), 256, 0, stream>>>(ctr2, sa3in, M3, 259, 288);
    maxpool_kernel<<<CDIV(M3 * 256, 256), 256, 0, stream>>>(XB, sa3in, M3, NS2, 256, 288, 3);

    // ---- SA3 (global) ----
    gemm(sa3in, wt_s3l1, fin[27], Yb, M3, 256, 288);
    bnblock(Yb, M3, 256, fin[28], fin[29], XA, 256, 0);
    gemm(XA, wt_s3l2, fin[31], Yb, M3, 512, 256);
    bnblock(Yb, M3, 512, fin[32], fin[33], XB, 512, 0);
    gemm(XB, wt_s3l3, fin[35], Yb, M3, 1024, 512);
    bnblock(Yb, M3, 1024, fin[36], fin[37], XA, 1024, 0);
    maxpool_kernel<<<CDIV(16 * 1024, 256), 256, 0, stream>>>(XA, Xh0, 16, S2, 1024, 1024, 0);

    // ---- head (M = 16, same WMMA GEMM + BN machinery) ----
    gemm(Xh0, wt_h1, fin[39], Yb, 16, 512, 1024);
    bnblock(Yb, 16, 512, fin[40], fin[41], Xh1, 512, 0);
    gemm(Xh1, wt_h2, fin[43], Yb, 16, 512, 512);
    bnblock(Yb, 16, 512, fin[44], fin[45], Xh2, 512, 0);
    gemm(Xh2, wt_h3a, nullptr, Yb, 16, 256, 512);
    bnblock(Yb, 16, 256, fin[47], fin[48], Xh3, 256, 1);
    gemm(Xh3, wt_h3b, nullptr, Yb, 16, 256, 256);
    bnblock(Yb, 16, 256, fin[50], fin[51], Xh4, 256, 1);
    gemm(Xh4, wt_h3c, nullptr, Yb, 16, 128, 256);
    bnblock(Yb, 16, 128, fin[53], fin[54], Xh5, 128, 1);

    head_final_kernel<<<1, 32, 0, stream>>>(Xh5, fin[55], fin[56], (float*)d_out);
}
